// TimeAwareCrossAttention_4767413699218
// MI455X (gfx1250) — compile-verified, matrix operand order checked by reading, not run on hardware
//
#include <hip/hip_runtime.h>
#include <cstdint>

// ---------------------------------------------------------------------------
// TimeAwareCrossAttention for MI455X (gfx1250, wave32, WMMA).
// Pipeline: f32->f16 conversion of activations/weights (once)
//           -> mask precompute
//           -> Q/K/V projections (f16 WMMA, f32 accum, raw fragment loads)
//           -> flash attention (async-LDS K staging, online softmax)
//           -> output projection.
// ---------------------------------------------------------------------------

typedef __attribute__((ext_vector_type(16))) _Float16 v16h;
typedef __attribute__((ext_vector_type(8)))  float    v8f;
typedef __attribute__((ext_vector_type(8)))  unsigned int v8u;
typedef int v4i_vs __attribute__((vector_size(16)));   // async builtin pointee

#define DIM 256
#define NH 8
#define HD 32
#define BB 8
#define LQ 512
#define TT 8192
#define MAXREL 128
#define ATTN_WAVES 4

#if __has_builtin(__builtin_amdgcn_global_load_async_to_lds_b128)
#define HAVE_ASYNC_LDS 1
typedef __attribute__((address_space(1))) v4i_vs* async_gptr_t;
typedef __attribute__((address_space(3))) v4i_vs* async_lptr_t;
#else
#define HAVE_ASYNC_LDS 0
#endif

// Starting K index (always even; the two halves of VGPR r are K0, K0+1) for
// the 16-bit 16x32 A layout (and the symmetric 32x16 B layout):
//   lanes 0-15 : V0..3 -> K 0..7,  V4..7 -> K 16..23
//   lanes 16-31: V0..3 -> K 8..15, V4..7 -> K 24..31
__device__ __forceinline__ int k0_of(int r, int hi) {
  int base = (r < 4) ? (2 * r) : (16 + 2 * (r - 4));
  return base + (hi ? 8 : 0);
}

__device__ __forceinline__ unsigned int pack2h(float x, float y) {
  union { _Float16 h[2]; unsigned int u; } p;
  p.h[0] = (_Float16)x; p.h[1] = (_Float16)y;
  return p.u;
}

// ---------------------------------------------------------------------------
// Kernel 0: bulk f32 -> f16 (8 elements/thread, b128 in / b128 out)
// ---------------------------------------------------------------------------
__global__ __launch_bounds__(256) void cvt_f16_kernel(
    const float* __restrict__ src, _Float16* __restrict__ dst, int n8) {
  int i = blockIdx.x * 256 + threadIdx.x;
  if (i >= n8) return;
  const float4* s = (const float4*)src + (size_t)i * 2;
  float4 a = s[0], b = s[1];
  uint4 o;
  o.x = pack2h(a.x, a.y);
  o.y = pack2h(a.z, a.w);
  o.z = pack2h(b.x, b.y);
  o.w = pack2h(b.z, b.w);
  *((uint4*)dst + i) = o;
}

// ---------------------------------------------------------------------------
// Kernel 1: relative-time mask, shared by all (b,h): mask[Lq][T]
// ---------------------------------------------------------------------------
__global__ __launch_bounds__(256) void mask_kernel(
    const float* __restrict__ rel_bias, const float* __restrict__ mask_scale,
    float* __restrict__ maskw) {
  int idx = blockIdx.x * 256 + threadIdx.x;
  if (idx >= LQ * TT) return;
  int qi = idx / TT, ti = idx % TT;          // TT power of 2 -> shifts
  float tau = (float)qi * ((float)(TT - 1) / (float)(LQ - 1));
  float dt = (float)ti - tau;
  float dtc = fminf(fmaxf(dt, -(float)MAXREL), (float)MAXREL);
  int bin = (int)dtc + MAXREL;               // trunc-toward-zero == astype(int32)
  float bias = rel_bias[bin];
  float u = dt * (1.0f / 64.0f);
  float lg = __logf(__expf(-0.5f * u * u) + 1e-6f);
  maskw[idx] = mask_scale[0] * (bias + lg);
}

// ---------------------------------------------------------------------------
// Kernel 2: projection  dst = (srch @ Wh[w_off:+256].T + b) * scale
// Sources are pre-converted f16 -> fragments are raw b32 loads (no cvt VALU).
// Each wave owns a 16x64 output strip: one A fragment feeds 4 WMMAs.
// Output: f16 head-major dst[((bb*NH + h)*RPB + t)*HD + hd].
// ---------------------------------------------------------------------------
template <int RPB>
__global__ __launch_bounds__(256) void proj_f16_kernel(
    const _Float16* __restrict__ srch, int nrows,
    const _Float16* __restrict__ Wh, const float* __restrict__ bias,
    int w_off, float scale, _Float16* __restrict__ dst) {
  const int lane = threadIdx.x & 31;
  const int wave = threadIdx.x >> 5;
  const int colGroups = DIM / 64;                 // 4 strips of 64 cols
  int tile = blockIdx.x * 8 + wave;
  int rowTiles = nrows / 16;
  if (tile >= rowTiles * colGroups) return;       // whole-wave exit: EXEC all-1
  int rowTile = tile / colGroups, colGroup = tile % colGroups;
  int row0 = rowTile * 16, o0 = colGroup * 64;
  const int hi = lane >> 4, lm = lane & 15;

  v8f acc[4] = {};
  const unsigned int* arow =
      (const unsigned int*)(srch + (size_t)(row0 + lm) * DIM);          // A: M = lm
  const unsigned int* brow =
      (const unsigned int*)(Wh + (size_t)(w_off + o0 + lm) * DIM);      // B: N = lm
  for (int dc = 0; dc < DIM / 32; ++dc) {
    int d0u = dc * 16;                            // u32 units (32 halves/step)
    v8u au;
#pragma unroll
    for (int r = 0; r < 8; ++r) au[r] = arow[d0u + (k0_of(r, hi) >> 1)];
    v16h a = __builtin_bit_cast(v16h, au);
#pragma unroll
    for (int j = 0; j < 4; ++j) {
      v8u bu;
#pragma unroll
      for (int r = 0; r < 8; ++r)
        bu[r] = brow[(size_t)(j * 16) * (DIM / 2) + d0u + (k0_of(r, hi) >> 1)];
      v16h b = __builtin_bit_cast(v16h, bu);
      acc[j] = __builtin_amdgcn_wmma_f32_16x16x32_f16(false, a, false, b,
                                                      (short)0, acc[j],
                                                      false, false);
    }
  }

#pragma unroll
  for (int j = 0; j < 4; ++j) {
    int o = o0 + j * 16 + lm;
    float bo = bias[w_off + o];
    int h = o >> 5, hd = o & 31;
#pragma unroll
    for (int r = 0; r < 8; ++r) {
      int n = row0 + r + hi * 8;                  // C/D: M = r + 8*hi
      int bb = n / RPB, t = n % RPB;              // shifts (RPB = 512 or 8192)
      size_t di = ((size_t)(bb * NH + h) * RPB + t) * HD + hd;
      dst[di] = (_Float16)((acc[j][r] + bo) * scale);
    }
  }
}

// ---------------------------------------------------------------------------
// Kernel 4 (launched last): out = attnh @ Wout.T + bout  (f32 result)
// ---------------------------------------------------------------------------
__global__ __launch_bounds__(256) void proj_out_kernel(
    const _Float16* __restrict__ srch, int nrows, const _Float16* __restrict__ Wh,
    const float* __restrict__ bias, float* __restrict__ dst) {
  const int lane = threadIdx.x & 31;
  const int wave = threadIdx.x >> 5;
  const int colGroups = DIM / 64;
  int tile = blockIdx.x * 8 + wave;
  int rowTiles = nrows / 16;
  if (tile >= rowTiles * colGroups) return;
  int rowTile = tile / colGroups, colGroup = tile % colGroups;
  int row0 = rowTile * 16, o0 = colGroup * 64;
  const int hi = lane >> 4, lm = lane & 15;

  v8f acc[4] = {};
  const unsigned int* arow =
      (const unsigned int*)(srch + (size_t)(row0 + lm) * DIM);
  const unsigned int* brow =
      (const unsigned int*)(Wh + (size_t)(o0 + lm) * DIM);
  for (int dc = 0; dc < DIM / 32; ++dc) {
    int d0u = dc * 16;
    v8u au;
#pragma unroll
    for (int r = 0; r < 8; ++r) au[r] = arow[d0u + (k0_of(r, hi) >> 1)];
    v16h a = __builtin_bit_cast(v16h, au);
#pragma unroll
    for (int j = 0; j < 4; ++j) {
      v8u bu;
#pragma unroll
      for (int r = 0; r < 8; ++r)
        bu[r] = brow[(size_t)(j * 16) * (DIM / 2) + d0u + (k0_of(r, hi) >> 1)];
      v16h b = __builtin_bit_cast(v16h, bu);
      acc[j] = __builtin_amdgcn_wmma_f32_16x16x32_f16(false, a, false, b,
                                                      (short)0, acc[j],
                                                      false, false);
    }
  }
#pragma unroll
  for (int j = 0; j < 4; ++j) {
    float bo = bias[o0 + j * 16 + lm];
#pragma unroll
    for (int r = 0; r < 8; ++r) {
      int n = row0 + r + hi * 8;
      dst[(size_t)n * DIM + o0 + j * 16 + lm] = acc[j][r] + bo;
    }
  }
}

// ---------------------------------------------------------------------------
// Kernel 3: flash attention. Block = 4 waves; each wave owns one 16-row q
// tile for a given (b,h). K/V streamed in 32-key chunks through LDS; K via
// GLOBAL_LOAD_ASYNC_TO_LDS_B128 (ASYNCcnt path), V manually transposed so PV
// B-fragments are contiguous b32 reads. Mask tile is the WMMA C initializer.
// Online softmax with shfl-xor(16) row reductions; P goes through per-wave
// LDS scratch (C-layout -> A-layout transpose). Output stored f16.
// ---------------------------------------------------------------------------
__global__ __launch_bounds__(ATTN_WAVES * 32) void attn_kernel(
    const _Float16* __restrict__ Qh, const _Float16* __restrict__ Kh,
    const _Float16* __restrict__ Vh, const float* __restrict__ maskw,
    _Float16* __restrict__ attn_out) {
  __shared__ unsigned short Ksm[32 * 32];                 // [key][hd]
  __shared__ unsigned short Vts[32 * 32];                 // [hd][key] transposed
  __shared__ unsigned short Psm[ATTN_WAVES][16 * 32];     // per-wave P [m][k]

  const int tid = threadIdx.x;
  const int lane = tid & 31, wave = tid >> 5;
  const int qBlocks = LQ / (16 * ATTN_WAVES);             // 8
  int bh = blockIdx.x / qBlocks;
  int qb = blockIdx.x % qBlocks;
  int b = bh / NH, h = bh % NH;
  int q0 = qb * (16 * ATTN_WAVES) + wave * 16;
  const int hi = lane >> 4, lm = lane & 15;

  // Q fragment (A layout), already pre-scaled by 1/sqrt(HD) at projection.
  v8u qu;
  {
    const unsigned int* qp =
        (const unsigned int*)(Qh + ((size_t)(b * NH + h) * LQ + q0 + lm) * HD);
#pragma unroll
    for (int r = 0; r < 8; ++r) qu[r] = qp[k0_of(r, hi) >> 1];
  }
  v16h qa = __builtin_bit_cast(v16h, qu);

  const _Float16* Kbase = Kh + (size_t)(b * NH + h) * TT * HD;
  const _Float16* Vbase = Vh + (size_t)(b * NH + h) * TT * HD;
  const float* mrow = maskw + (size_t)q0 * TT;

  v8f o0acc = {}, o1acc = {};
  float mrun[8], lrun[8];
#pragma unroll
  for (int r = 0; r < 8; ++r) { mrun[r] = -3.0e38f; lrun[r] = 0.0f; }

  for (int t0 = 0; t0 < TT; t0 += 32) {
    __syncthreads();  // previous chunk's LDS reads done before overwrite
    {
      // cooperative stage: 128 threads x 16B covers one 32x32 f16 chunk each
      int key = tid >> 2;
      int hb = (tid & 3) * 8;
#if HAVE_ASYNC_LDS
      // K: direct global->LDS async copy (no VGPR round trip, ASYNCcnt)
      __builtin_amdgcn_global_load_async_to_lds_b128(
          (async_gptr_t)(Kbase + (size_t)(t0 + key) * HD + hb),
          (async_lptr_t)&Ksm[key * 32 + hb],
          0, 0);
#else
      uint4 kd = *(const uint4*)(Kbase + (size_t)(t0 + key) * HD + hb);
      *(uint4*)&Ksm[key * 32 + hb] = kd;
#endif
      // V: through VGPRs, stored transposed
      uint4 vd = *(const uint4*)(Vbase + (size_t)(t0 + key) * HD + hb);
      const unsigned short* vs = (const unsigned short*)&vd;
#pragma unroll
      for (int i = 0; i < 8; ++i) Vts[(hb + i) * 32 + key] = vs[i];
      if (t0 + 32 < TT) {
        __builtin_prefetch(Kbase + (size_t)(t0 + 32 + key) * HD + hb, 0, 1);
        __builtin_prefetch(Vbase + (size_t)(t0 + 32 + key) * HD + hb, 0, 1);
      }
#if HAVE_ASYNC_LDS
      asm volatile("s_wait_asynccnt 0" ::: "memory");
#endif
    }
    __syncthreads();

    // S = mask + Q*K^T : mask tile loaded straight into the C operand
    v8f s0, s1;
#pragma unroll
    for (int r = 0; r < 8; ++r) {
      const float* mr = mrow + (size_t)(r + hi * 8) * TT + t0 + lm;
      s0[r] = mr[0];
      s1[r] = mr[16];
    }
    {
      v8u b0u, b1u;
#pragma unroll
      for (int r = 0; r < 8; ++r) {
        int k0 = k0_of(r, hi);                      // k0 = hd pair (contiguous)
        b0u[r] = *(const unsigned int*)&Ksm[lm * 32 + k0];         // keys 0..15
        b1u[r] = *(const unsigned int*)&Ksm[(lm + 16) * 32 + k0];  // keys 16..31
      }
      v16h bk0 = __builtin_bit_cast(v16h, b0u);
      v16h bk1 = __builtin_bit_cast(v16h, b1u);
      s0 = __builtin_amdgcn_wmma_f32_16x16x32_f16(false, qa, false, bk0,
                                                  (short)0, s0, false, false);
      s1 = __builtin_amdgcn_wmma_f32_16x16x32_f16(false, qa, false, bk1,
                                                  (short)0, s1, false, false);
    }

    // online softmax (row M lives across one 16-lane half per VGPR r)
#pragma unroll
    for (int r = 0; r < 8; ++r) {
      float mx = fmaxf(s0[r], s1[r]);
#pragma unroll
      for (int d = 1; d < 16; d <<= 1) mx = fmaxf(mx, __shfl_xor(mx, d, 16));
      float mnew = fmaxf(mrun[r], mx);
      float alpha = __expf(mrun[r] - mnew);
      float p0 = __expf(s0[r] - mnew);
      float p1 = __expf(s1[r] - mnew);
      float rs = p0 + p1;
#pragma unroll
      for (int d = 1; d < 16; d <<= 1) rs += __shfl_xor(rs, d, 16);
      lrun[r] = lrun[r] * alpha + rs;
      mrun[r] = mnew;
      o0acc[r] *= alpha;
      o1acc[r] *= alpha;
      // stash P in C-layout positions for the LDS transpose
      int m = r + hi * 8;
      union { _Float16 hh; unsigned short us; } c0, c1;
      c0.hh = (_Float16)p0; c1.hh = (_Float16)p1;
      Psm[wave][m * 32 + lm] = c0.us;
      Psm[wave][m * 32 + lm + 16] = c1.us;
    }
    // same-wave DS is in-order; wait makes the RAW through LDS airtight
    asm volatile("s_wait_dscnt 0" ::: "memory");

    // O += P[16x32] * V[32xHD]  (two 16-wide hd tiles)
    {
      v8u pu, v0u, v1u;
#pragma unroll
      for (int r = 0; r < 8; ++r) {
        int k0 = k0_of(r, hi);                      // k0 = key pair here
        pu[r]  = *(const unsigned int*)&Psm[wave][lm * 32 + k0];
        v0u[r] = *(const unsigned int*)&Vts[lm * 32 + k0];          // hd 0..15
        v1u[r] = *(const unsigned int*)&Vts[(lm + 16) * 32 + k0];   // hd 16..31
      }
      v16h pa  = __builtin_bit_cast(v16h, pu);
      v16h vb0 = __builtin_bit_cast(v16h, v0u);
      v16h vb1 = __builtin_bit_cast(v16h, v1u);
      o0acc = __builtin_amdgcn_wmma_f32_16x16x32_f16(false, pa, false, vb0,
                                                     (short)0, o0acc,
                                                     false, false);
      o1acc = __builtin_amdgcn_wmma_f32_16x16x32_f16(false, pa, false, vb1,
                                                     (short)0, o1acc,
                                                     false, false);
    }
  }

  // normalize and store attn_out[b][q][h*HD + hd] (f16, DIM-major)
#pragma unroll
  for (int r = 0; r < 8; ++r) {
    float inv = 1.0f / lrun[r];
    int q = q0 + r + hi * 8;
    _Float16* op = attn_out + ((size_t)(b * LQ) + q) * DIM + h * HD;
    op[lm]      = (_Float16)(o0acc[r] * inv);
    op[lm + 16] = (_Float16)(o1acc[r] * inv);
  }
}

// ---------------------------------------------------------------------------
// Host launcher
// ---------------------------------------------------------------------------
extern "C" void kernel_launch(void* const* d_in, const int* in_sizes, int n_in,
                              void* d_out, int out_size, void* d_ws, size_t ws_size,
                              hipStream_t stream) {
  (void)in_sizes; (void)n_in; (void)out_size; (void)ws_size;
  const float* q          = (const float*)d_in[0];
  const float* kv         = (const float*)d_in[1];
  const float* in_w       = (const float*)d_in[2];
  const float* in_b       = (const float*)d_in[3];
  const float* out_w      = (const float*)d_in[4];
  const float* out_b      = (const float*)d_in[5];
  const float* rel_bias   = (const float*)d_in[6];
  const float* mask_scale = (const float*)d_in[7];

  char* ws = (char*)d_ws;
  size_t off = 0;
  auto carve = [&](size_t bytes) {
    char* p = ws + off;
    off += (bytes + 255) & ~(size_t)255;
    return p;
  };
  _Float16* kvh   = (_Float16*)carve((size_t)BB * TT * DIM * 2);    // 32 MB
  _Float16* qh    = (_Float16*)carve((size_t)BB * LQ * DIM * 2);    //  2 MB
  _Float16* inwh  = (_Float16*)carve((size_t)3 * DIM * DIM * 2);    // 384 KB
  _Float16* outwh = (_Float16*)carve((size_t)DIM * DIM * 2);        // 128 KB
  _Float16* Qh    = (_Float16*)carve((size_t)BB * LQ * DIM * 2);    //  2 MB
  _Float16* Kh    = (_Float16*)carve((size_t)BB * TT * DIM * 2);    // 32 MB
  _Float16* Vh    = (_Float16*)carve((size_t)BB * TT * DIM * 2);    // 32 MB
  float*    mskw  = (float*)carve((size_t)LQ * TT * 4);             // 16 MB
  _Float16* attnh = (_Float16*)carve((size_t)BB * LQ * DIM * 2);    //  2 MB

  // 0) one-time f32 -> f16 conversions (kills per-tile cvt VALU, halves reads)
  {
    int n8 = BB * TT * DIM / 8;
    cvt_f16_kernel<<<(n8 + 255) / 256, 256, 0, stream>>>(kv, kvh, n8);
  }
  {
    int n8 = BB * LQ * DIM / 8;
    cvt_f16_kernel<<<(n8 + 255) / 256, 256, 0, stream>>>(q, qh, n8);
  }
  {
    int n8 = 3 * DIM * DIM / 8;
    cvt_f16_kernel<<<(n8 + 255) / 256, 256, 0, stream>>>(in_w, inwh, n8);
  }
  {
    int n8 = DIM * DIM / 8;
    cvt_f16_kernel<<<(n8 + 255) / 256, 256, 0, stream>>>(out_w, outwh, n8);
  }
  // 1) mask (shared by all b,h; stays resident in the 192MB L2)
  {
    int n = LQ * TT;
    mask_kernel<<<(n + 255) / 256, 256, 0, stream>>>(rel_bias, mask_scale, mskw);
  }
  // 2) projections (Q pre-scaled by 1/sqrt(HD))
  const float qscale = 0.17677669529663687f;  // 1/sqrt(32)
  {
    int tiles = (BB * LQ / 16) * (DIM / 64);
    proj_f16_kernel<LQ><<<(tiles + 7) / 8, 256, 0, stream>>>(
        qh, BB * LQ, inwh, in_b, 0, qscale, Qh);
  }
  {
    int tiles = (BB * TT / 16) * (DIM / 64);
    proj_f16_kernel<TT><<<(tiles + 7) / 8, 256, 0, stream>>>(
        kvh, BB * TT, inwh, in_b, DIM, 1.0f, Kh);
    proj_f16_kernel<TT><<<(tiles + 7) / 8, 256, 0, stream>>>(
        kvh, BB * TT, inwh, in_b, 2 * DIM, 1.0f, Vh);
  }
  // 3) flash attention: 512 blocks x 128 threads (4 waves)
  attn_kernel<<<BB * NH * (LQ / (16 * ATTN_WAVES)), ATTN_WAVES * 32, 0, stream>>>(
      Qh, Kh, Vh, mskw, attnh);
  // 4) output projection -> f32 d_out
  {
    int tiles = (BB * LQ / 16) * (DIM / 64);
    proj_out_kernel<<<(tiles + 7) / 8, 256, 0, stream>>>(
        attnh, BB * LQ, outwh, out_b, (float*)d_out);
  }
}
